// MamMIL_39754217292333
// MI455X (gfx1250) — matrix operand order repaired and checked by reading, not compiled, not run on hardware
//
#include <hip/hip_runtime.h>
#include <hip/hip_bf16.h>
#include <stdint.h>

#define HID 512

typedef __bf16 v16bf __attribute__((ext_vector_type(16)));
typedef __bf16 v8bf  __attribute__((ext_vector_type(8)));
typedef float  v8f   __attribute__((ext_vector_type(8)));

union Frag { v16bf v; v8bf h[2]; };

// ---------------------------------------------------------------------------
// GEMM: out[M,N] = act(A[M,K] @ W[N,K]^T + bias[N]); fp32 in/out, bf16 WMMA.
// Block = 256 threads (8 wave32), tile 64(M) x 128(N), K chunked by 32 in LDS.
// ---------------------------------------------------------------------------
#define KC 32
#define A_STRIDE 40   // bf16 elems per LDS row (32 + 8 pad) = 80B, 16B aligned
#define MT 64
#define NTB 128

__device__ __forceinline__ void st_bf16x4(__bf16* d, float4 f) {
  d[0] = (__bf16)f.x; d[1] = (__bf16)f.y; d[2] = (__bf16)f.z; d[3] = (__bf16)f.w;
}

__global__ __launch_bounds__(256) void gemm_relu_wmma(
    const float* __restrict__ A, const float* __restrict__ W,
    const float* __restrict__ bias, float* __restrict__ out,
    int M, int K, int N, int do_relu)
{
  extern __shared__ char smem[];
  __bf16* lds_a = (__bf16*)smem;                       // [64][A_STRIDE]
  __bf16* lds_w = (__bf16*)(smem + MT * A_STRIDE * 2); // [128][A_STRIDE]

  const int tid  = threadIdx.x;
  const int lane = tid & 31;
  const int wave = tid >> 5;
  const int m0 = blockIdx.x * MT;
  const int n0 = blockIdx.y * NTB;

  // wave tile: 2 mtiles x 2 ntiles of 16x16
  const int wm = (wave >> 2) * 2;   // mtile base 0 / 2
  const int wn = (wave & 3) * 2;    // ntile base 0,2,4,6

  v8f acc[2][2] = {};

  const int half = lane >> 4;       // wave32 lane halves
  const int mrow = lane & 15;

  // per-thread staging slots (fixed across K iterations)
  int ra0 = tid >> 3,           ca0 = (tid & 7) * 4;           // A slot 0
  int ra1 = (tid + 256) >> 3,   ca1 = ca0;                     // A slot 1
  int ga0 = m0 + ra0; if (ga0 >= M) ga0 = M - 1;
  int ga1 = m0 + ra1; if (ga1 >= M) ga1 = M - 1;
  const float* pa0 = A + (size_t)ga0 * K + ca0;
  const float* pa1 = A + (size_t)ga1 * K + ca1;
  // W slots: 4 rows, stride 32 rows apart
  const float* pw0 = W + (size_t)(n0 + ra0) * K + ca0;
  const float* pw1 = W + (size_t)(n0 + ra0 + 32) * K + ca0;
  const float* pw2 = W + (size_t)(n0 + ra0 + 64) * K + ca0;
  const float* pw3 = W + (size_t)(n0 + ra0 + 96) * K + ca0;

  __bf16* da0 = lds_a + ra0 * A_STRIDE + ca0;
  __bf16* da1 = lds_a + ra1 * A_STRIDE + ca1;
  __bf16* dw0 = lds_w + ra0 * A_STRIDE + ca0;
  __bf16* dw1 = lds_w + (ra0 + 32) * A_STRIDE + ca0;
  __bf16* dw2 = lds_w + (ra0 + 64) * A_STRIDE + ca0;
  __bf16* dw3 = lds_w + (ra0 + 96) * A_STRIDE + ca0;

  for (int kb = 0; kb < K; kb += KC) {
    // issue ALL staging loads into independent temps (batch the latency),
    // then convert + store to LDS.
    const float4 fa0 = *(const float4*)(pa0 + kb);
    const float4 fa1 = *(const float4*)(pa1 + kb);
    const float4 fw0 = *(const float4*)(pw0 + kb);
    const float4 fw1 = *(const float4*)(pw1 + kb);
    const float4 fw2 = *(const float4*)(pw2 + kb);
    const float4 fw3 = *(const float4*)(pw3 + kb);
    st_bf16x4(da0, fa0);
    st_bf16x4(da1, fa1);
    st_bf16x4(dw0, fw0);
    st_bf16x4(dw1, fw1);
    st_bf16x4(dw2, fw2);
    st_bf16x4(dw3, fw3);

    // prefetch next K chunk while this one is consumed (global_prefetch_b8)
    if (kb + KC < K) {
      __builtin_prefetch(pa0 + kb + KC, 0, 1);
      __builtin_prefetch(pw0 + kb + KC, 0, 1);
      __builtin_prefetch(pw2 + kb + KC, 0, 1);
    }
    __syncthreads();

    // A frag (16x32): lanes 0-15 rows, K = {8h..8h+7, 16+8h..16+8h+7}
    Frag afr[2], bfr[2];
    #pragma unroll
    for (int i = 0; i < 2; i++) {
      const __bf16* ar = lds_a + ((wm + i) * 16 + mrow) * A_STRIDE;
      afr[i].h[0] = *(const v8bf*)(ar + 8 * half);
      afr[i].h[1] = *(const v8bf*)(ar + 16 + 8 * half);
    }
    // B frag (32x16): lane n = lane%16, K = {8h..8h+7, 16+8h..16+8h+7}
    #pragma unroll
    for (int j = 0; j < 2; j++) {
      const __bf16* br = lds_w + ((wn + j) * 16 + (lane & 15)) * A_STRIDE;
      bfr[j].h[0] = *(const v8bf*)(br + 8 * half);
      bfr[j].h[1] = *(const v8bf*)(br + 16 + 8 * half);
    }
    #pragma unroll
    for (int i = 0; i < 2; i++)
      #pragma unroll
      for (int j = 0; j < 2; j++)
        acc[i][j] = __builtin_amdgcn_wmma_f32_16x16x32_bf16(
            false, afr[i].v, false, bfr[j].v, (short)0, acc[i][j], false, false);
    __syncthreads();
  }

  // epilogue: D layout — VGPR v: M = v + 8*half; N = lane%16
  const bool full_tile = (m0 + MT <= M);   // only last M-block is partial
  #pragma unroll
  for (int i = 0; i < 2; i++) {
    int gmbase = m0 + (wm + i) * 16 + 8 * half;
    #pragma unroll
    for (int j = 0; j < 2; j++) {
      int gn = n0 + (wn + j) * 16 + (lane & 15);
      float b = bias[gn];
      if (full_tile) {
        float* p = out + (size_t)gmbase * N + gn;
        #pragma unroll
        for (int v = 0; v < 8; v++) {
          float x = acc[i][j][v] + b;
          if (do_relu) x = x > 0.f ? x : 0.f;
          *p = x;
          p += N;
        }
      } else {
        #pragma unroll
        for (int v = 0; v < 8; v++) {
          int gm = gmbase + v;
          if (gm < M) {
            float x = acc[i][j][v] + b;
            if (do_relu) x = x > 0.f ? x : 0.f;
            out[(size_t)gm * N + gn] = x;
          }
        }
      }
    }
  }
}

// ---------------------------------------------------------------------------
// Distributed GRU scan: 8 WGs, WG g owns hidden units [64g, 64g+64).
// Whh rows {j, 512+j, 1024+j} resident in LDS as bf16 (~195 KB).
// Global atomic barrier (monotonic counter) + double-buffered h per step.
// ---------------------------------------------------------------------------
#define SCAN_WGS 8
#define WROWS 192          // 3 gates * 64 hidden units
#define WSTRIDE 520        // 512 + 8 pad (1040B rows, 16B aligned)

__global__ __launch_bounds__(256) void gru_scan(
    const float* __restrict__ Whh, const float* __restrict__ bhh,
    const float* __restrict__ gi, float* __restrict__ h0buf,
    float* __restrict__ h1buf, unsigned int* __restrict__ counter, int N)
{
  extern __shared__ char smem[];
  __bf16* lds_w   = (__bf16*)smem;                          // [192][WSTRIDE]
  float*  lds_h   = (float*)(smem + WROWS * WSTRIDE * 2);   // [512]
  float*  lds_acc = lds_h + HID;                            // [192]
  float*  lds_b   = lds_acc + WROWS;                        // [192]

  const int tid = threadIdx.x;
  const int g = blockIdx.x;
  const int jbase = g * 64;

  // resident Whh slice: local row r = part*64 + jj  <->  global row part*512 + jbase + jj
  for (int e = tid; e < WROWS * (HID / 4); e += 256) {
    int r = e / (HID / 4), c4 = e % (HID / 4);
    int part = r >> 6, jj = r & 63;
    const float4 f = *(const float4*)(Whh + (size_t)(part * HID + jbase + jj) * HID + c4 * 4);
    st_bf16x4(lds_w + r * WSTRIDE + c4 * 4, f);
  }
  if (tid < WROWS) {
    int part = tid >> 6, jj = tid & 63;
    lds_b[tid] = bhh[part * HID + jbase + jj];
  }
  __syncthreads();

  for (int t = 0; t < N; t++) {
    const float* hin  = (t & 1) ? h1buf : h0buf;
    float*       hout = (t & 1) ? h0buf : h1buf;
    lds_h[tid]       = hin[tid];
    lds_h[tid + 256] = hin[tid + 256];
    __syncthreads();

    if (tid < WROWS) {
      // bf16 row dot fp32 h; unpack bf16 via bit ops (no cvt needed)
      const uint4* wr = (const uint4*)(lds_w + (size_t)tid * WSTRIDE);
      float acc = 0.f;
      #pragma unroll 4
      for (int k8 = 0; k8 < HID / 8; k8++) {
        uint4 q = wr[k8];
        const float* hp = lds_h + k8 * 8;
        acc += __uint_as_float((q.x & 0xffffu) << 16) * hp[0];
        acc += __uint_as_float( q.x & 0xffff0000u)    * hp[1];
        acc += __uint_as_float((q.y & 0xffffu) << 16) * hp[2];
        acc += __uint_as_float( q.y & 0xffff0000u)    * hp[3];
        acc += __uint_as_float((q.z & 0xffffu) << 16) * hp[4];
        acc += __uint_as_float( q.z & 0xffff0000u)    * hp[5];
        acc += __uint_as_float((q.w & 0xffffu) << 16) * hp[6];
        acc += __uint_as_float( q.w & 0xffff0000u)    * hp[7];
      }
      lds_acc[tid] = acc;
    } else if (t + 1 < N) {
      // idle lanes: prefetch next step's gi row (global_prefetch_b8)
      int p = tid - WROWS;            // 0..63, 48 cachelines cover 1536 f32
      if (p < 48)
        __builtin_prefetch(gi + (size_t)(t + 1) * (3 * HID) + p * 32, 0, 1);
    }
    __syncthreads();

    if (tid < 64) {
      int j = jbase + tid;
      const float* git = gi + (size_t)t * (3 * HID);
      float ir = git[j], iz = git[HID + j], inn = git[2 * HID + j];
      float hr = lds_acc[tid]       + lds_b[tid];
      float hz = lds_acc[64 + tid]  + lds_b[64 + tid];
      float hn = lds_acc[128 + tid] + lds_b[128 + tid];
      float r = 1.f / (1.f + __expf(-(ir + hr)));
      float u = 1.f / (1.f + __expf(-(iz + hz)));
      float n = tanhf(inn + r * hn);
      hout[j] = (1.f - u) * n + u * lds_h[j];
    }
    __threadfence();
    __syncthreads();
    if (tid == 0) {
      __hip_atomic_fetch_add(counter, 1u, __ATOMIC_RELEASE, __HIP_MEMORY_SCOPE_AGENT);
      unsigned target = (unsigned)(t + 1) * SCAN_WGS;
      while (__hip_atomic_load(counter, __ATOMIC_ACQUIRE, __HIP_MEMORY_SCOPE_AGENT) < target)
        __builtin_amdgcn_s_sleep(2);
    }
    __syncthreads();
  }
}

__global__ void scan_init(float* h0, float* h1, unsigned int* counter) {
  int i = threadIdx.x;               // 512 threads
  h0[i] = 0.f; h1[i] = 0.f;
  if (i == 0) *counter = 0u;
}

__global__ __launch_bounds__(256) void readout(
    const float* __restrict__ Wr, const float* __restrict__ br,
    const float* __restrict__ hT, float* __restrict__ out)
{
  __shared__ float hs[HID];
  int tid = threadIdx.x;
  hs[tid] = hT[tid]; hs[tid + 256] = hT[tid + 256];
  __syncthreads();
  int f = blockIdx.x * 256 + tid;
  const float* w = Wr + (size_t)f * HID;
  float acc = br[f];
  #pragma unroll 8
  for (int k = 0; k < HID; k++) acc += w[k] * hs[k];
  out[f] = acc;
}

extern "C" void kernel_launch(void* const* d_in, const int* in_sizes, int n_in,
                              void* d_out, int out_size, void* d_ws, size_t ws_size,
                              hipStream_t stream) {
  const float* features = (const float*)d_in[0];
  const float* Wp  = (const float*)d_in[1];
  const float* bp  = (const float*)d_in[2];
  const float* Wih = (const float*)d_in[3];
  const float* Whh = (const float*)d_in[4];
  const float* bih = (const float*)d_in[5];
  const float* bhh = (const float*)d_in[6];
  const float* Wr  = (const float*)d_in[7];
  const float* br  = (const float*)d_in[8];

  const int Np = in_sizes[0] / HID;   // 100000
  float* out = (float*)d_out;

  size_t off = 0;
  float* z  = (float*)((char*)d_ws + off); off += (size_t)Np * HID * sizeof(float);
  float* gi = (float*)((char*)d_ws + off); off += (size_t)Np * 3 * HID * sizeof(float);
  off = (off + 255) & ~(size_t)255;
  float* h0 = (float*)((char*)d_ws + off); off += 2048;
  float* h1 = (float*)((char*)d_ws + off); off += 2048;
  unsigned int* counter = (unsigned int*)((char*)d_ws + off);

  int Mb = (Np + MT - 1) / MT;
  size_t gemm_lds = (size_t)(MT + NTB) * A_STRIDE * 2;   // bf16 staging

  // z = relu(features @ Wp^T + bp)
  gemm_relu_wmma<<<dim3(Mb, HID / NTB), 256, gemm_lds, stream>>>(
      features, Wp, bp, z, Np, HID, HID, 1);
  // gi = z @ Wih^T + bih
  gemm_relu_wmma<<<dim3(Mb, (3 * HID) / NTB), 256, gemm_lds, stream>>>(
      z, Wih, bih, gi, Np, HID, 3 * HID, 0);

  scan_init<<<1, 512, 0, stream>>>(h0, h1, counter);

  size_t scan_lds = (size_t)WROWS * WSTRIDE * 2 + (HID + WROWS + WROWS) * sizeof(float);
  gru_scan<<<SCAN_WGS, 256, scan_lds, stream>>>(Whh, bhh, gi, h0, h1, counter, Np);

  const float* hT = (Np & 1) ? h1 : h0;   // step t writes (t&1)?h0:h1
  readout<<<HID / 256, 256, 0, stream>>>(Wr, br, hT, out);
}
